// GammaGRUCRF_24541443130174
// MI455X (gfx1250) — matrix operand order repaired and checked by reading, not compiled, not run on hardware
//
#include <hip/hip_runtime.h>

typedef unsigned short u16;
typedef __attribute__((ext_vector_type(4)))  unsigned int u32x4;
typedef __attribute__((ext_vector_type(16))) __bf16       v16bf;
typedef __attribute__((ext_vector_type(8)))  float        v8f;
typedef int v4i __attribute__((vector_size(16)));   // matches async-LDS builtin param

union Frag {
  v16bf v;
  u32x4 q[2];
};

constexpr int B = 64, T = 512, E = 128, H = 256, K = 16;
constexpr int G3 = 3 * H;   // 768 gate width
constexpr int BT = B * T;   // 32768 rows
constexpr int H2 = 2 * H;   // 512

#if defined(__gfx1250__) && __has_builtin(__builtin_amdgcn_global_load_async_to_lds_b128)
#define GRU_ASYNC 1
#else
#define GRU_ASYNC 0
#endif

__device__ __forceinline__ u16 f2bf(float f) {
  unsigned u = __float_as_uint(f);
  unsigned r = u + 0x7FFFu + ((u >> 16) & 1u);   // round-to-nearest-even
  return (u16)(r >> 16);
}
__device__ __forceinline__ float bf2f(u16 h) {
  return __uint_as_float(((unsigned)h) << 16);
}
__device__ __forceinline__ float sigmoidf(float x) {
  return 1.0f / (1.0f + __expf(-x));
}

// ---------------------------------------------------------------- utilities
__global__ __launch_bounds__(256) void cvt_kernel(const float* __restrict__ src,
                                                  u16* __restrict__ dst, int n) {
  int i = blockIdx.x * 256 + threadIdx.x;
  if (i < n) dst[i] = f2bf(src[i]);
}

__global__ __launch_bounds__(256) void embed_kernel(const int* __restrict__ x,
                                                    const float* __restrict__ ew,
                                                    u16* __restrict__ xe) {
  const size_t i = (size_t)blockIdx.x * 256 + threadIdx.x;   // over BT*E
  const int    e = (int)(i & (E - 1));
  const size_t bt = i >> 7;                                  // E == 128
  const int tok = x[bt];
  xe[i] = f2bf(ew[(size_t)tok * E + e]);
}

// ------------------------------------------------- xp = xe @ w_ih^T + b_ih
__global__ __launch_bounds__(256) void xp_gemm_kernel(const u16* __restrict__ xe,
                                                      const u16* __restrict__ wih,
                                                      const float* __restrict__ bih,
                                                      float* __restrict__ xp) {
  const int lane = threadIdx.x & 31;
  const int wave = threadIdx.x >> 5;
  const int tile = blockIdx.x * 8 + wave;        // (BT/16)*(G3/16) tiles
  const int mT = tile / (G3 / 16);
  const int nT = tile % (G3 / 16);
  const int g = lane >> 4, mn = lane & 15;

  const u16* aRow = xe  + (size_t)(mT * 16 + mn) * E;
  const u16* bRow = wih + (size_t)(nT * 16 + mn) * E;

  v8f acc = {};
#pragma unroll
  for (int kk = 0; kk < E; kk += 32) {
    Frag fa, fb;
    fa.q[0] = *(const u32x4*)(aRow + kk + g * 8);
    fa.q[1] = *(const u32x4*)(aRow + kk + 16 + g * 8);
    fb.q[0] = *(const u32x4*)(bRow + kk + g * 8);
    fb.q[1] = *(const u32x4*)(bRow + kk + 16 + g * 8);
    acc = __builtin_amdgcn_wmma_f32_16x16x32_bf16(false, fa.v, false, fb.v,
                                                  (short)0, acc, false, false);
  }
  const float bias = bih[nT * 16 + mn];
#pragma unroll
  for (int r = 0; r < 8; ++r) {
    const int row = mT * 16 + r + g * 8;
    xp[(size_t)row * G3 + nT * 16 + mn] = acc[r] + bias;
  }
}

// ----------------------------------------------------------- GRU recurrence
// Grid: 8 blocks = 2 dirs x 4 batch-groups of 16; 512 threads (16 waves).
// h (16x256 bf16), gh (16x768 f32) and the per-step xp slice (16x768 f32,
// async-prefetched) live in LDS; per step 48 WMMA N-tiles over 16 waves.
__global__ __launch_bounds__(512) void gru_kernel(
    const float* __restrict__ xp_f, const float* __restrict__ xp_b,
    const u16* __restrict__ whh_f, const u16* __restrict__ whh_b,
    const float* __restrict__ bhh_f, const float* __restrict__ bhh_b,
    u16* __restrict__ h_out) {
  __shared__ __align__(16) u16 h_sh[16 * H];   // 8 KB
  __shared__ float gh_sh[16 * G3];             // 48 KB
#if GRU_ASYNC
  __shared__ __align__(16) float xp_sh[16 * G3];  // 48 KB async staging
#endif

  const int tid = threadIdx.x;
  const int dir = blockIdx.x >> 2;
  const int bg  = blockIdx.x & 3;
  const int lane = tid & 31, wave = tid >> 5;
  const int g = lane >> 4, mn = lane & 15;

  const float* xp  = dir ? xp_b  : xp_f;
  const u16*   whh = dir ? whh_b : whh_f;
  const float* bhh = dir ? bhh_b : bhh_f;

  for (int i = tid; i < 16 * H; i += 512) h_sh[i] = 0;   // h0 = 0
  __syncthreads();

  for (int s = 0; s < T; ++s) {
    const int tt = dir ? (T - 1 - s) : s;

#if GRU_ASYNC
    // Kick the async DMA of this step's xp slice (16 rows x 768 f32) into
    // LDS; 3072 x 16B transfers hidden behind the WMMA phase below.
#pragma unroll
    for (int p = 0; p < 6; ++p) {
      const int idx = tid + p * 512;      // 0..3071
      const int bb = idx / 192;           // row in group (768 f32 = 192 x 16B)
      const int c  = idx % 192;           // 16B chunk within row
      const float* gp = xp + ((size_t)(bg * 16 + bb) * T + tt) * G3 + c * 4;
      float* lp = &xp_sh[bb * G3 + c * 4];
      __builtin_amdgcn_global_load_async_to_lds_b128(
          (__attribute__((address_space(1))) v4i*)gp,
          (__attribute__((address_space(3))) v4i*)lp, 0, 0);
    }
#endif

    // ---- gh = h @ whh^T + bhh : 48 N-tiles over 16 waves, K = 256 ----
#pragma unroll
    for (int it = 0; it < 3; ++it) {
      const int nT = wave * 3 + it;
      const u16* bRow = whh + (size_t)(nT * 16 + mn) * H;
      const u16* aRow = h_sh + mn * H;
      __builtin_prefetch(bRow, 0, 1);
      v8f acc = {};
#pragma unroll
      for (int kk = 0; kk < H; kk += 32) {
        Frag fa, fb;
        fa.q[0] = *(const u32x4*)(aRow + kk + g * 8);        // ds_load_b128
        fa.q[1] = *(const u32x4*)(aRow + kk + 16 + g * 8);
        fb.q[0] = *(const u32x4*)(bRow + kk + g * 8);
        fb.q[1] = *(const u32x4*)(bRow + kk + 16 + g * 8);
        acc = __builtin_amdgcn_wmma_f32_16x16x32_bf16(false, fa.v, false, fb.v,
                                                      (short)0, acc, false, false);
      }
      const int n = nT * 16 + mn;
      const float bias = bhh[n];
#pragma unroll
      for (int r = 0; r < 8; ++r) gh_sh[(r + g * 8) * G3 + n] = acc[r] + bias;
    }

#if GRU_ASYNC
#if __has_builtin(__builtin_amdgcn_s_wait_asynccnt)
    __builtin_amdgcn_s_wait_asynccnt(0);
#else
    asm volatile("s_wait_asynccnt 0x0" ::: "memory");
#endif
#endif
    __syncthreads();

    // ---- elementwise gates: 16*256 values, 8 per thread ----
#pragma unroll
    for (int u = 0; u < 8; ++u) {
      const int idx = tid * 8 + u;
      const int bb = idx >> 8;       // 0..15 batch row in group
      const int j  = idx & 255;      // hidden index
#if GRU_ASYNC
      const float xr = xp_sh[bb * G3 + j];
      const float xz = xp_sh[bb * G3 + H + j];
      const float xn = xp_sh[bb * G3 + 2 * H + j];
#else
      const size_t xrow = ((size_t)(bg * 16 + bb) * T + tt) * G3;
      const float xr = xp[xrow + j];
      const float xz = xp[xrow + H + j];
      const float xn = xp[xrow + 2 * H + j];
#endif
      const float hr = gh_sh[bb * G3 + j];
      const float hz = gh_sh[bb * G3 + H + j];
      const float hn = gh_sh[bb * G3 + 2 * H + j];
      const float r  = sigmoidf(xr + hr);
      const float z  = sigmoidf(xz + hz);
      const float nn = tanhf(xn + r * hn);
      const float hp = bf2f(h_sh[bb * H + j]);
      const float hv = (1.0f - z) * nn + z * hp;
      const u16 hb = f2bf(hv);
      h_sh[bb * H + j] = hb;
      h_out[((size_t)(bg * 16 + bb) * T + tt) * H2 + dir * H + j] = hb;
    }
    __syncthreads();
  }
}

// ------------------------------------------------- em = h @ fc_w^T + fc_b
__global__ __launch_bounds__(256) void fc_kernel(const u16* __restrict__ h,
                                                 const u16* __restrict__ fcw,
                                                 const float* __restrict__ fcb,
                                                 float* __restrict__ em) {
  const int lane = threadIdx.x & 31;
  const int wave = threadIdx.x >> 5;
  const int mT = blockIdx.x * 8 + wave;          // 2048 row tiles, 1 col tile
  const int g = lane >> 4, mn = lane & 15;

  const u16* aRow = h   + (size_t)(mT * 16 + mn) * H2;
  const u16* bRow = fcw + (size_t)mn * H2;       // col n = row n of fc_w

  v8f acc = {};
#pragma unroll 4
  for (int kk = 0; kk < H2; kk += 32) {
    Frag fa, fb;
    fa.q[0] = *(const u32x4*)(aRow + kk + g * 8);
    fa.q[1] = *(const u32x4*)(aRow + kk + 16 + g * 8);
    fb.q[0] = *(const u32x4*)(bRow + kk + g * 8);
    fb.q[1] = *(const u32x4*)(bRow + kk + 16 + g * 8);
    acc = __builtin_amdgcn_wmma_f32_16x16x32_bf16(false, fa.v, false, fb.v,
                                                  (short)0, acc, false, false);
  }
  const float bias = fcb[mn];
#pragma unroll
  for (int r = 0; r < 8; ++r)
    em[(size_t)(mT * 16 + r + g * 8) * K + mn] = acc[r] + bias;
}

// ------------------------------------------------------------ CRF Viterbi
__global__ __launch_bounds__(32) void viterbi_kernel(const float* __restrict__ em,
                                                     const int* __restrict__ y,
                                                     const float* __restrict__ trans,
                                                     float* __restrict__ out) {
  __shared__ unsigned char bp_sh[(T - 1) * K];
  const int b = blockIdx.x;
  const int lane = threadIdx.x;
  const int j = lane & 15;

  float tcol[K];
#pragma unroll
  for (int i = 0; i < K; ++i) tcol[i] = trans[i * K + j];   // column j

  float score = trans[1 * K + j] + em[(size_t)b * T * K + j];   // BOS = 1
  for (int t = 1; t < T; ++t) {
    const float e = em[((size_t)b * T + t) * K + j];
    const bool m = (y[b * T + t] != 0);                        // PAD = 0
    float best = -__builtin_inff();
    int bi = 0;
#pragma unroll
    for (int i = 0; i < K; ++i) {
      const float si = __shfl(score, i, 32);
      const float c = si + tcol[i];
      if (c > best) { best = c; bi = i; }                      // first-max
    }
    int bpv = j;                                               // identity
    if (m) { score = best + e; bpv = bi; }
    if (lane < 16) bp_sh[(t - 1) * K + j] = (unsigned char)bpv;
  }
  score += trans[j * K + 2];                                   // EOS col
  float best = -__builtin_inff();
  int tag = 0;
#pragma unroll
  for (int i = 0; i < K; ++i) {
    const float si = __shfl(score, i, 32);
    if (si > best) { best = si; tag = i; }
  }
  __syncthreads();                                             // 1-wave: S_NOP
  if (lane == 0) {
    out[b] = best;
    float* path = out + B + (size_t)b * T;
    int tg = tag;
    path[T - 1] = (y[b * T + T - 1] != 0) ? (float)tg : 0.0f;
    for (int t = T - 2; t >= 0; --t) {
      const int prev = bp_sh[t * K + tg];
      path[t] = (y[b * T + t] != 0) ? (float)prev : 0.0f;
      tg = prev;
    }
  }
}

// ------------------------------------------------------------------ launch
extern "C" void kernel_launch(void* const* d_in, const int* in_sizes, int n_in,
                              void* d_out, int out_size, void* d_ws, size_t ws_size,
                              hipStream_t stream) {
  (void)in_sizes; (void)n_in; (void)out_size; (void)ws_size;
  const int*   x       = (const int*)  d_in[0];
  const int*   y       = (const int*)  d_in[1];
  const float* embed_w = (const float*)d_in[2];
  const float* w_ih_f  = (const float*)d_in[3];
  const float* w_hh_f  = (const float*)d_in[4];
  const float* b_ih_f  = (const float*)d_in[5];
  const float* b_hh_f  = (const float*)d_in[6];
  const float* w_ih_b  = (const float*)d_in[7];
  const float* w_hh_b  = (const float*)d_in[8];
  const float* b_ih_b  = (const float*)d_in[9];
  const float* b_hh_b  = (const float*)d_in[10];
  const float* fc_w    = (const float*)d_in[11];
  const float* fc_b    = (const float*)d_in[12];
  const float* trans   = (const float*)d_in[13];
  float* out = (float*)d_out;

  char* base = (char*)d_ws;
  size_t off = 0;
  auto take = [&](size_t bytes) -> char* {
    off = (off + 255) & ~(size_t)255;
    char* p = base + off;
    off += bytes;
    return p;
  };
  u16*   xe      = (u16*)  take((size_t)BT * E * 2);    //  8   MB
  u16*   wihf_bf = (u16*)  take((size_t)G3 * E * 2);
  u16*   wihb_bf = (u16*)  take((size_t)G3 * E * 2);
  u16*   whhf_bf = (u16*)  take((size_t)G3 * H * 2);
  u16*   whhb_bf = (u16*)  take((size_t)G3 * H * 2);
  u16*   fcw_bf  = (u16*)  take((size_t)K * H2 * 2);
  float* xpf     = (float*)take((size_t)BT * G3 * 4);   // 96   MB
  float* xpb     = (float*)take((size_t)BT * G3 * 4);   // 96   MB
  u16*   hout    = (u16*)  take((size_t)BT * H2 * 2);   // 32   MB
  float* em      = (float*)take((size_t)BT * K * 4);    //  2   MB

  cvt_kernel<<<(G3 * E + 255) / 256, 256, 0, stream>>>(w_ih_f, wihf_bf, G3 * E);
  cvt_kernel<<<(G3 * E + 255) / 256, 256, 0, stream>>>(w_ih_b, wihb_bf, G3 * E);
  cvt_kernel<<<(G3 * H + 255) / 256, 256, 0, stream>>>(w_hh_f, whhf_bf, G3 * H);
  cvt_kernel<<<(G3 * H + 255) / 256, 256, 0, stream>>>(w_hh_b, whhb_bf, G3 * H);
  cvt_kernel<<<(K * H2 + 255) / 256, 256, 0, stream>>>(fc_w, fcw_bf, K * H2);

  embed_kernel<<<(BT * E) / 256, 256, 0, stream>>>(x, embed_w, xe);

  const int xpBlocks = (BT / 16) * (G3 / 16) / 8;   // 12288 blocks, 8 waves each
  xp_gemm_kernel<<<xpBlocks, 256, 0, stream>>>(xe, wihf_bf, b_ih_f, xpf);
  xp_gemm_kernel<<<xpBlocks, 256, 0, stream>>>(xe, wihb_bf, b_ih_b, xpb);

  gru_kernel<<<8, 512, 0, stream>>>(xpf, xpb, whhf_bf, whhb_bf, b_hh_f, b_hh_b, hout);

  fc_kernel<<<(BT / 16) / 8, 256, 0, stream>>>(hout, fcw_bf, fc_b, em);

  viterbi_kernel<<<B, 32, 0, stream>>>(em, y, trans, out);
}
